// Decoder_42064909697280
// MI455X (gfx1250) — compile-verified, matrix operand order checked by reading, not run on hardware
//
#include <hip/hip_runtime.h>
#include <hip/hip_bf16.h>

// ---------------- Problem constants (from reference) ----------------
#define B_    128
#define SDEC_ 256
#define SENC_ 256
#define H_    1024
#define V_    256
#define O_    256
#define DIN_  1280          // V + H
#define H3_   3072          // 3*H

typedef __bf16 bf16_t;
typedef __attribute__((ext_vector_type(8)))  __bf16 v8bf;
typedef __attribute__((ext_vector_type(16))) __bf16 v16bf;
typedef __attribute__((ext_vector_type(8)))  float  v8f;

// ---------------------------------------------------------------
// Fragment loader for v_wmma_f32_16x16x32_bf16.
// ISA 16-bit A-matrix layout (16x32, MxK): lane = 16*hi + lo holds
//   row M = lo, K in [hi*8, hi*8+8) for VGPR0-3 and [16+hi*8, ...) for VGPR4-7,
// i.e. two contiguous 8-element (16-byte) chunks per lane. Weights are stored
// with K contiguous per output row so B fragments use the symmetric layout
// (lane -> N row of W, same K chunking).
// ---------------------------------------------------------------
__device__ __forceinline__ v16bf load_frag16x32(const bf16_t* __restrict__ p,
                                                int row_stride, int row, int k, int hi) {
  const bf16_t* r = p + (size_t)row * row_stride + k + hi * 8;
  v8bf x = *(const v8bf*)(r);
  v8bf y = *(const v8bf*)(r + 16);
  return __builtin_shufflevector(x, y, 0,1,2,3,4,5,6,7,8,9,10,11,12,13,14,15);
}

// ---------------------------------------------------------------
// Core 32x32-per-wave GEMM body: C[M x N] = A[M x K] * W^T (+ bias).
// Block = 128 threads = 4 waves arranged 2x2; block tile = 64(M) x 64(N).
// Each wave holds 4 f32 accumulators (2 M-tiles x 2 N-tiles) so both A and B
// fragments are reused 2x per K-step -> 4 WMMAs per 4 fragment loads.
// ldc allows scattering rows (used to write logits straight into d_out).
// ---------------------------------------------------------------
__device__ __forceinline__
void gemm32_body(const bf16_t* __restrict__ A, int lda,
                 const bf16_t* __restrict__ W, int ldw,
                 float* __restrict__ Cf, bf16_t* __restrict__ Cb,
                 long long ldc, const float* __restrict__ bias,
                 int M, int N, int K) {
  const int wave = threadIdx.x >> 5;
  const int lane = threadIdx.x & 31;
  const int lo = lane & 15;
  const int hi = lane >> 4;
  const int m0 = blockIdx.y * 64 + (wave >> 1) * 32;
  const int n0 = blockIdx.x * 64 + (wave & 1) * 32;
  if (m0 >= M || n0 >= N) return;

  const v8f zero = {0.f, 0.f, 0.f, 0.f, 0.f, 0.f, 0.f, 0.f};
  v8f acc00 = zero, acc01 = zero, acc10 = zero, acc11 = zero;

  for (int k = 0; k < K; k += 32) {
    v16bf a0 = load_frag16x32(A, lda, m0 + lo,      k, hi);
    v16bf a1 = load_frag16x32(A, lda, m0 + 16 + lo, k, hi);
    v16bf b0 = load_frag16x32(W, ldw, n0 + lo,      k, hi);
    v16bf b1 = load_frag16x32(W, ldw, n0 + 16 + lo, k, hi);
    acc00 = __builtin_amdgcn_wmma_f32_16x16x32_bf16(false, a0, false, b0, (short)0, acc00, false, false);
    acc01 = __builtin_amdgcn_wmma_f32_16x16x32_bf16(false, a0, false, b1, (short)0, acc01, false, false);
    acc10 = __builtin_amdgcn_wmma_f32_16x16x32_bf16(false, a1, false, b0, (short)0, acc10, false, false);
    acc11 = __builtin_amdgcn_wmma_f32_16x16x32_bf16(false, a1, false, b1, (short)0, acc11, false, false);
  }

  // f32 C/D layout: VGPR i -> M = mt + i + 8*hi ; lane lo -> N column.
  v8f accs[2][2] = {{acc00, acc01}, {acc10, acc11}};
#pragma unroll
  for (int mt = 0; mt < 2; ++mt) {
    const int rbase = m0 + mt * 16 + hi * 8;
#pragma unroll
    for (int nt = 0; nt < 2; ++nt) {
      const int n = n0 + nt * 16 + lo;
      const float bv = bias ? bias[n] : 0.0f;
#pragma unroll
      for (int i = 0; i < 8; ++i) {
        const float val = accs[mt][nt][i] + bv;
        const size_t idx = (size_t)(rbase + i) * (size_t)ldc + (size_t)n;
        if (Cf) Cf[idx] = val;
        if (Cb) Cb[idx] = (bf16_t)val;
      }
    }
  }
}

__global__ __launch_bounds__(128)
void gemm_wmma_bf16(const bf16_t* __restrict__ A, int lda,
                    const bf16_t* __restrict__ W, int ldw,
                    float* __restrict__ Cf, bf16_t* __restrict__ Cb,
                    long long ldc, const float* __restrict__ bias,
                    int M, int N, int K) {
  gemm32_body(A, lda, W, ldw, Cf, Cb, ldc, bias, M, N, K);
}

// Two independent GEMMs in one launch (blockIdx.z selects the operand set).
// Used for [gi | gh] of a GRU layer and for [logits(t) | q(t+1)].
__global__ __launch_bounds__(128)
void gemm_wmma_bf16_dual(const bf16_t* __restrict__ A0, int lda0,
                         const bf16_t* __restrict__ W0, int ldw0,
                         float* __restrict__ Cf0, long long ldc0,
                         const float* __restrict__ bias0, int N0, int K0,
                         const bf16_t* __restrict__ A1, int lda1,
                         const bf16_t* __restrict__ W1, int ldw1,
                         float* __restrict__ Cf1, long long ldc1,
                         const float* __restrict__ bias1, int N1, int K1,
                         int M) {
  if (blockIdx.z == 0)
    gemm32_body(A0, lda0, W0, ldw0, Cf0, nullptr, ldc0, bias0, M, N0, K0);
  else
    gemm32_body(A1, lda1, W1, ldw1, Cf1, nullptr, ldc1, bias1, M, N1, K1);
}

// ---------------- setup / conversion kernels ----------------
__global__ void cvt_f32_bf16(const float* __restrict__ s, bf16_t* __restrict__ d, size_t n) {
  size_t i = (size_t)blockIdx.x * 256 + threadIdx.x;
  if (i < n) d[i] = (bf16_t)s[i];
}

// attn_W is (2H x H) row-major; Wq = attn_W[:H], Wk = attn_W[H:].
// We need Wq^T / Wk^T stored as (output row j, K=h contiguous) for the GEMM.
__global__ void build_attnW(const float* __restrict__ attn_W,
                            bf16_t* __restrict__ wqT, bf16_t* __restrict__ wkT) {
  size_t i = (size_t)blockIdx.x * 256 + threadIdx.x;
  if (i >= (size_t)H_ * H_) return;
  int j = (int)(i / H_);
  int h = (int)(i % H_);
  wqT[i] = (bf16_t)attn_W[(size_t)h * H_ + j];
  wkT[i] = (bf16_t)attn_W[(size_t)(H_ + h) * H_ + j];
}

__global__ void init_h(const float* __restrict__ enc_hidden,
                       float* __restrict__ h0f, float* __restrict__ h1f,
                       bf16_t* __restrict__ h0b, bf16_t* __restrict__ h1b) {
  int i = blockIdx.x * 256 + threadIdx.x;
  if (i >= B_ * H_) return;
  float a = enc_hidden[i];
  float b = enc_hidden[(size_t)B_ * H_ + i];
  h0f[i] = a; h1f[i] = b;
  h0b[i] = (bf16_t)a; h1b[i] = (bf16_t)b;
}

// ---------------------------------------------------------------
// Fused attention: one block per batch row.
//  scores[s] = sum_h tanh(q[h] + encpb[s,h]) * v[h]  (encpb = enc@Wk + attn_b, bf16)
//  softmax over S, context[h] = sum_s w[s]*enc[s,h],
//  rnn_in = [x_t | context] in bf16.
// ---------------------------------------------------------------
__global__ __launch_bounds__(256)
void attn_context_kernel(const float* __restrict__ q,
                         const bf16_t* __restrict__ encpb,
                         const bf16_t* __restrict__ enc_bf,
                         const float* __restrict__ attn_v,
                         const bf16_t* __restrict__ dec_bf,
                         int t,
                         bf16_t* __restrict__ rnn_in) {
  const int b   = blockIdx.x;
  const int tid = threadIdx.x;
  __shared__ float qv[H_];
  __shared__ float vv[H_];
  __shared__ float red[SENC_];
  __shared__ float wsm[SENC_];

  for (int h = tid; h < H_; h += 256) {
    qv[h] = q[(size_t)b * H_ + h];
    vv[h] = attn_v[h];
  }
  __syncthreads();

  const int wave = tid >> 5;
  const int lane = tid & 31;
  const size_t ebase = (size_t)b * SENC_ * H_;

  for (int s = wave; s < SENC_; s += 8) {
    const bf16_t* ep = encpb + ebase + (size_t)s * H_;
    float acc = 0.0f;
    for (int h = lane; h < H_; h += 32)
      acc += tanhf(qv[h] + (float)ep[h]) * vv[h];
    for (int off = 16; off > 0; off >>= 1) acc += __shfl_xor(acc, off, 32);
    if (lane == 0) wsm[s] = acc;
  }
  __syncthreads();

  // softmax over 256 scores (blockDim == SENC_)
  float sc = wsm[tid];
  red[tid] = sc; __syncthreads();
  for (int st = 128; st > 0; st >>= 1) {
    if (tid < st) red[tid] = fmaxf(red[tid], red[tid + st]);
    __syncthreads();
  }
  const float mx = red[0]; __syncthreads();
  const float e = __expf(sc - mx);
  red[tid] = e; __syncthreads();
  for (int st = 128; st > 0; st >>= 1) {
    if (tid < st) red[tid] += red[tid + st];
    __syncthreads();
  }
  const float w = e / red[0];
  __syncthreads();
  wsm[tid] = w;
  __syncthreads();

  // context -> rnn_in[V..V+H)
  for (int h = tid; h < H_; h += 256) {
    float acc = 0.0f;
    const bf16_t* ec = enc_bf + ebase + h;
    for (int s = 0; s < SENC_; ++s) acc += wsm[s] * (float)ec[(size_t)s * H_];
    rnn_in[(size_t)b * DIN_ + V_ + h] = (bf16_t)acc;
  }
  // x_t -> rnn_in[0..V)
  for (int i = tid; i < V_; i += 256)
    rnn_in[(size_t)b * DIN_ + i] = dec_bf[((size_t)b * SDEC_ + t) * V_ + i];
}

// ---------------- GRU gate math (in-place hidden update) ----------------
__global__ void gru_gate(const float* __restrict__ gi, const float* __restrict__ gh,
                         const float* __restrict__ bi, const float* __restrict__ bh,
                         const float* __restrict__ hprev,
                         float* __restrict__ hnew, bf16_t* __restrict__ hnewb) {
  int idx = blockIdx.x * 256 + threadIdx.x;
  if (idx >= B_ * H_) return;
  int b = idx >> 10;          // / H_
  int n = idx & (H_ - 1);     // % H_
  size_t base = (size_t)b * H3_;
  float ir  = gi[base + n]           + bi[n];
  float iz  = gi[base + H_ + n]      + bi[H_ + n];
  float in_ = gi[base + 2 * H_ + n]  + bi[2 * H_ + n];
  float hr  = gh[base + n]           + bh[n];
  float hz  = gh[base + H_ + n]      + bh[H_ + n];
  float hn  = gh[base + 2 * H_ + n]  + bh[2 * H_ + n];
  float r = 1.0f / (1.0f + __expf(-(ir + hr)));
  float z = 1.0f / (1.0f + __expf(-(iz + hz)));
  float nn = tanhf(in_ + r * hn);
  float out = (1.0f - z) * nn + z * hprev[idx];
  hnew[idx]  = out;
  hnewb[idx] = (bf16_t)out;
}

// ---------------------------------------------------------------
extern "C" void kernel_launch(void* const* d_in, const int* in_sizes, int n_in,
                              void* d_out, int out_size, void* d_ws, size_t ws_size,
                              hipStream_t stream) {
  const float* dec_inputs = (const float*)d_in[0];
  const float* enc_hidden = (const float*)d_in[1];
  const float* enc_outs   = (const float*)d_in[2];
  const float* attn_W     = (const float*)d_in[3];
  const float* attn_b     = (const float*)d_in[4];
  const float* attn_v     = (const float*)d_in[5];
  const float* Wi0        = (const float*)d_in[6];
  const float* Wh0        = (const float*)d_in[7];
  const float* bi0        = (const float*)d_in[8];
  const float* bh0        = (const float*)d_in[9];
  const float* Wi1        = (const float*)d_in[10];
  const float* Wh1        = (const float*)d_in[11];
  const float* bi1        = (const float*)d_in[12];
  const float* bh1        = (const float*)d_in[13];
  const float* out_W      = (const float*)d_in[14];
  const float* out_b      = (const float*)d_in[15];
  float* out = (float*)d_out;

  // ---- workspace bump allocator (256B aligned) ----
  char* ws = (char*)d_ws;
  size_t off = 0;
  auto alloc = [&](size_t bytes) -> void* {
    void* p = ws + off;
    off = (off + bytes + 255) & ~(size_t)255;
    return p;
  };

  bf16_t* wqT    = (bf16_t*)alloc((size_t)H_ * H_ * 2);
  bf16_t* wkT    = (bf16_t*)alloc((size_t)H_ * H_ * 2);
  bf16_t* wi0b   = (bf16_t*)alloc((size_t)H3_ * DIN_ * 2);
  bf16_t* wh0b   = (bf16_t*)alloc((size_t)H3_ * H_ * 2);
  bf16_t* wi1b   = (bf16_t*)alloc((size_t)H3_ * H_ * 2);
  bf16_t* wh1b   = (bf16_t*)alloc((size_t)H3_ * H_ * 2);
  bf16_t* owb    = (bf16_t*)alloc((size_t)O_ * H_ * 2);
  bf16_t* enc_bf = (bf16_t*)alloc((size_t)B_ * SENC_ * H_ * 2);
  bf16_t* dec_bf = (bf16_t*)alloc((size_t)B_ * SDEC_ * V_ * 2);
  bf16_t* encpb  = (bf16_t*)alloc((size_t)B_ * SENC_ * H_ * 2);
  float*  qbuf   = (float*)alloc((size_t)B_ * H_ * 4);
  bf16_t* rnn_in = (bf16_t*)alloc((size_t)B_ * DIN_ * 2);
  float*  gi     = (float*)alloc((size_t)B_ * H3_ * 4);
  float*  gh     = (float*)alloc((size_t)B_ * H3_ * 4);
  float*  h0f    = (float*)alloc((size_t)B_ * H_ * 4);
  float*  h1f    = (float*)alloc((size_t)B_ * H_ * 4);
  bf16_t* h0b    = (bf16_t*)alloc((size_t)B_ * H_ * 2);
  bf16_t* h1b    = (bf16_t*)alloc((size_t)B_ * H_ * 2);
  (void)ws_size; (void)in_sizes; (void)n_in; (void)out_size;

  auto cvt = [&](const float* s, bf16_t* d, size_t n) {
    cvt_f32_bf16<<<(unsigned)((n + 255) / 256), 256, 0, stream>>>(s, d, n);
  };

  // ---- one-time (per call) conversions ----
  cvt(Wi0, wi0b, (size_t)H3_ * DIN_);
  cvt(Wh0, wh0b, (size_t)H3_ * H_);
  cvt(Wi1, wi1b, (size_t)H3_ * H_);
  cvt(Wh1, wh1b, (size_t)H3_ * H_);
  cvt(out_W, owb, (size_t)O_ * H_);
  cvt(enc_outs, enc_bf, (size_t)B_ * SENC_ * H_);
  cvt(dec_inputs, dec_bf, (size_t)B_ * SDEC_ * V_);
  build_attnW<<<(unsigned)(((size_t)H_ * H_ + 255) / 256), 256, 0, stream>>>(attn_W, wqT, wkT);
  init_h<<<(B_ * H_) / 256, 256, 0, stream>>>(enc_hidden, h0f, h1f, h0b, h1b);

  // enc_proj + attn_b -> bf16 (M = B*SENC, N = H, K = H); lives in L2 across steps.
  gemm_wmma_bf16<<<dim3(H_ / 64, (B_ * SENC_) / 64), 128, 0, stream>>>(
      enc_bf, H_, wkT, H_, nullptr, encpb, (long long)H_, attn_b,
      B_ * SENC_, H_, H_);

  // prologue: q for t = 0
  gemm_wmma_bf16<<<dim3(H_ / 64, B_ / 64), 128, 0, stream>>>(
      h1b, H_, wqT, H_, qbuf, nullptr, (long long)H_, nullptr, B_, H_, H_);

  const dim3 gg(H3_ / 64, B_ / 64, 2);   // [gi | gh]
  const dim3 glq(H_ / 64, B_ / 64, 2);   // [logits | q(t+1)]  (logits blocks self-guard on N)

  // ---- sequential decode over 256 steps ----
  for (int t = 0; t < SDEC_; ++t) {
    // attention + rnn_in assembly (consumes qbuf from previous fused launch)
    attn_context_kernel<<<B_, 256, 0, stream>>>(qbuf, encpb, enc_bf, attn_v,
                                                dec_bf, t, rnn_in);
    // GRU layer 0: gi = rnn_in@Wi0^T, gh = h0@Wh0^T (independent -> one launch)
    gemm_wmma_bf16_dual<<<gg, 128, 0, stream>>>(
        rnn_in, DIN_, wi0b, DIN_, gi, (long long)H3_, nullptr, H3_, DIN_,
        h0b,    H_,   wh0b, H_,   gh, (long long)H3_, nullptr, H3_, H_,
        B_);
    gru_gate<<<(B_ * H_) / 256, 256, 0, stream>>>(gi, gh, bi0, bh0, h0f, h0f, h0b);
    // GRU layer 1
    gemm_wmma_bf16_dual<<<gg, 128, 0, stream>>>(
        h0b, H_, wi1b, H_, gi, (long long)H3_, nullptr, H3_, H_,
        h1b, H_, wh1b, H_, gh, (long long)H3_, nullptr, H3_, H_,
        B_);
    gru_gate<<<(B_ * H_) / 256, 256, 0, stream>>>(gi, gh, bi1, bh1, h1f, h1f, h1b);
    // logits(t) scattered into d_out[(b*SDEC + t)*O + o]  +  q(t+1) = h1@Wq
    gemm_wmma_bf16_dual<<<glq, 128, 0, stream>>>(
        h1b, H_, owb, H_, out + (size_t)t * O_, (long long)SDEC_ * O_, out_b, O_, H_,
        h1b, H_, wqT, H_, qbuf,                 (long long)H_,         nullptr, H_, H_,
        B_);
  }
}
